// LBKT_21320217657955
// MI455X (gfx1250) — matrix operand order repaired and checked by reading, not compiled, not run on hardware
//
#include <hip/hip_runtime.h>

typedef __attribute__((ext_vector_type(16))) __bf16 v16bf;
typedef __attribute__((ext_vector_type(8)))  float  v8f;

#define BB   32
#define LL   100
#define CC   128
#define HH   128

__device__ __forceinline__ float sigf(float x) { return 1.0f / (1.0f + __expf(-x)); }

// ---------------------------------------------------------------------------
// Kernel 1: pack W_in and W1 (= gate3_W rows 0:128) into WMMA B-fragment order,
// and precompute column sums of gate3_W rows 256:306 / 306:356 / 356:406.
// B-fragment layout (bf16 32x16): lanes 0-15: N=lane, e0-7 -> K 0-7, e8-15 -> K 16-23;
// lanes 16-31: N=lane-16, e0-7 -> K 8-15, e8-15 -> K 24-31 (plus 32*kb).
// ---------------------------------------------------------------------------
__global__ __launch_bounds__(256) void lbkt_pack(
    const float* __restrict__ W_in, const float* __restrict__ gate3_W,
    __bf16* __restrict__ winp, __bf16* __restrict__ w1p, float* __restrict__ wsums)
{
    int tid = threadIdx.x;
    // W_in pack: [kb(8)][n(8)][lane(32)][e(16)]
    for (int i = tid; i < 8 * 8 * 32 * 16; i += 256) {
        int e = i & 15, ln = (i >> 4) & 31, n = (i >> 9) & 7, kb = i >> 12;
        int N = n * 16 + (ln & 15);
        int K = kb * 32 + (e & 7) + ((e >= 8) ? 16 : 0) + ((ln >= 16) ? 8 : 0);
        winp[i] = (__bf16)W_in[K * HH + N];
    }
    // W1 pack: [n(8)][kb(4)][lane(32)][e(16)]
    for (int i = tid; i < 8 * 4 * 32 * 16; i += 256) {
        int e = i & 15, ln = (i >> 4) & 31, kb = (i >> 9) & 3, n = i >> 11;
        int N = n * 16 + (ln & 15);
        int K = kb * 32 + (e & 7) + ((e >= 8) ? 16 : 0) + ((ln >= 16) ? 8 : 0);
        w1p[i] = (__bf16)gate3_W[K * HH + N];
    }
    // column sums for the broadcast factor rows (t: 256:306, a: 306:356, h: 356:406)
    for (int i = tid; i < 3 * HH; i += 256) {
        int which = i >> 7, h = i & 127;
        int r0 = 256 + which * 50;
        float s = 0.f;
        for (int r = 0; r < 50; ++r) s += gate3_W[(r0 + r) * HH + h];
        wsums[i] = s;
    }
}

// ---------------------------------------------------------------------------
// Kernel 2: acts = relu([E_q[q], E_c[c]] @ W_in + b_in)  -> [3200, 128]
// M=3200 (25 blocks x 8 waves x 16 rows), K=256 (8 kblocks), N=128 (8 ntiles).
// A-fragment (bf16 16x32): lanes 0-15: M=lane, e0-7 -> K 0-7, e8-15 -> K 16-23;
// lanes 16-31: M=lane-16, e0-7 -> K 8-15, e8-15 -> K 24-31.
// ---------------------------------------------------------------------------
__global__ __launch_bounds__(256) void lbkt_acts(
    const int* __restrict__ qseq, const int* __restrict__ cseq,
    const float* __restrict__ E_q, const float* __restrict__ E_c,
    const __bf16* __restrict__ winp, const float* __restrict__ b_in,
    float* __restrict__ acts)
{
    int tid = threadIdx.x, lane = tid & 31, wave = tid >> 5;
    int strip = blockIdx.x * 8 + wave;            // 0..199
    int mrow = strip * 16 + (lane & 15);
    int qi = qseq[mrow], ci = cseq[mrow];

    v8f acc[8];
    v8f z = {0.f, 0.f, 0.f, 0.f, 0.f, 0.f, 0.f, 0.f};
#pragma unroll
    for (int n = 0; n < 8; ++n) acc[n] = z;

#pragma unroll
    for (int kb = 0; kb < 8; ++kb) {
        int base = (kb & 3) * 32 + ((lane >= 16) ? 8 : 0);
        const float* src = (kb < 4) ? (E_q + (size_t)qi * 128 + base)
                                    : (E_c + (size_t)ci * 128 + base);
        float4 r0 = *(const float4*)(src);
        float4 r1 = *(const float4*)(src + 4);
        float4 r2 = *(const float4*)(src + 16);
        float4 r3 = *(const float4*)(src + 20);
        v16bf a;
        a[0] = (__bf16)r0.x;  a[1] = (__bf16)r0.y;  a[2] = (__bf16)r0.z;  a[3] = (__bf16)r0.w;
        a[4] = (__bf16)r1.x;  a[5] = (__bf16)r1.y;  a[6] = (__bf16)r1.z;  a[7] = (__bf16)r1.w;
        a[8] = (__bf16)r2.x;  a[9] = (__bf16)r2.y;  a[10] = (__bf16)r2.z; a[11] = (__bf16)r2.w;
        a[12] = (__bf16)r3.x; a[13] = (__bf16)r3.y; a[14] = (__bf16)r3.z; a[15] = (__bf16)r3.w;
#pragma unroll
        for (int n = 0; n < 8; ++n) {
            v16bf bf = *(const v16bf*)(winp + (((kb * 8 + n) * 32 + lane) << 4));
            acc[n] = __builtin_amdgcn_wmma_f32_16x16x32_bf16(
                false, a, false, bf, (short)0, acc[n], false, false);
        }
    }
    int rbase = strip * 16 + ((lane >= 16) ? 8 : 0);
    int cl = lane & 15;
#pragma unroll
    for (int n = 0; n < 8; ++n) {
        int col = n * 16 + cl;
        float bb = b_in[col];
#pragma unroll
        for (int v = 0; v < 8; ++v) {
            float val = acc[n][v] + bb;
            acts[(size_t)(rbase + v) * HH + col] = val > 0.f ? val : 0.f;
        }
    }
}

// ---------------------------------------------------------------------------
// Kernel 3: the recurrent scan. One workgroup (8 wave32) per batch, persistent
// over 100 steps. h kept f32 in LDS + bf16 fragment-packed shadow for WMMA.
// ---------------------------------------------------------------------------
__global__ __launch_bounds__(256) void lbkt_scan(
    const int* __restrict__ qseq,
    const float* __restrict__ tf, const float* __restrict__ af, const float* __restrict__ hf,
    const float* __restrict__ E_q, const float* __restrict__ q_matrix,
    const float* __restrict__ init_h,
    const float* __restrict__ tg_W, const float* __restrict__ tg_b,
    const float* __restrict__ ag_W, const float* __restrict__ ag_b,
    const float* __restrict__ hg_W, const float* __restrict__ hg_b,
    const float* __restrict__ timeW, const float* __restrict__ attW, const float* __restrict__ hintW,
    const float* __restrict__ Wf, const float* __restrict__ cell_bias,
    const float* __restrict__ gate3_W, const float* __restrict__ gate3_b,
    const float* __restrict__ out_W, const float* __restrict__ out_b,
    const float* __restrict__ acts, const __bf16* __restrict__ w1p_g,
    const float* __restrict__ wsums, float* __restrict__ out)
{
    extern __shared__ char smem[];
    float*  hS   = (float*)smem;                          // 128x128 f32 (64 KB)
    __bf16* hA   = (__bf16*)(smem + 65536);               // A-frag packed  (32 KB)
    __bf16* W1p  = (__bf16*)(smem + 65536 + 32768);       // B-frag packed  (32 KB)
    float*  dyn  = (float*)(smem + 131072);
    float* corrS = dyn;            // 128
    float* topicS = dyn + 128;     // 128
    float* interS = dyn + 256;     // 128
    float* htil  = dyn + 384;      // 128
    float* predv = dyn + 512;      // 128
    float* gateS = dyn + 640;      // 3*128
    float* fus   = dyn + 1024;     // 3*4*128
    float* baseS = dyn + 2560;     // 128
    float* lgS   = dyn + 2688;     // 128

    int b = blockIdx.x, tid = threadIdx.x, lane = tid & 31, wave = tid >> 5;

    // init state: f32 copy + bf16 fragment-packed copy of broadcast init_h
    for (int i = tid; i < CC * HH; i += 256) {
        float v = init_h[i];
        hS[i] = v;
        int c = i >> 7, hc = i & 127;
        int strip = c >> 4, kb = hc >> 5, r = hc & 31;
        int lg2 = (r >> 3) & 1;
        int e = (r & 7) + ((r >= 16) ? 8 : 0);
        hA[(((strip * 4 + kb) * 32 + ((c & 15) + 16 * lg2)) << 4) + e] = (__bf16)v;
    }
    // copy pre-packed W1 fragments into LDS
    for (int i = tid; i < (8 * 4 * 32 * 16) / 4; i += 256)
        ((unsigned long long*)W1p)[i] = ((const unsigned long long*)w1p_g)[i];
    __syncthreads();

    const float* wt = wsums;
    const float* wa = wsums + 128;
    const float* wh = wsums + 256;

    for (int l = 0; l < LL; ++l) {
        int idx = b * LL + l;
        int q = qseq[idx];
        float tF = tf[idx], aF = af[idx], hF = hf[idx];
        if (tid < 128) {
            corrS[tid]  = q_matrix[(size_t)q * CC + tid];
            topicS[tid] = E_q[(size_t)q * 128 + tid];
            interS[tid] = acts[(size_t)idx * HH + tid];
        }
        __syncthreads();

        // phase A: h_tilde[h] = sum_c corr[c] * h[c][h]
        if (tid < 128) {
            float s = 0.f;
#pragma unroll 8
            for (int c = 0; c < CC; ++c) s += corrS[c] * hS[c * HH + tid];
            htil[tid] = s;
        }
        __syncthreads();

        // phase B: 4 gemvs ([htil,x2] @ W), 512 jobs
        for (int j = tid; j < 512; j += 256) {
            int gm = j >> 7, h = j & 127;
            const float *W, *bb, *x2; float fac = 0.f;
            if (gm == 0)      { W = out_W; bb = out_b; x2 = topicS; }
            else if (gm == 1) { W = tg_W;  bb = tg_b;  x2 = interS; fac = tF; }
            else if (gm == 2) { W = ag_W;  bb = ag_b;  x2 = interS; fac = aF; }
            else              { W = hg_W;  bb = hg_b;  x2 = interS; fac = hF; }
            float s = bb[h];
#pragma unroll 8
            for (int k = 0; k < 128; ++k) s += htil[k] * W[k * HH + h];
#pragma unroll 8
            for (int k = 0; k < 128; ++k) s += x2[k] * W[(128 + k) * HH + h];
            if (gm == 0) {
                predv[h] = sigf(s);
            } else {
                // swapped-arg gate (torch bug replicated): 0.7 + 0.3*sigmoid(fac-0.3)
                float g = 0.7f + 0.3f * sigf(fac - 0.3f);
                gateS[(gm - 1) * 128 + h] = sigf(s * g);
            }
        }
        __syncthreads();

        // phase C: fusion gemvs (jobs 0..383) + forget base vector (384..511)
        for (int j = tid; j < 512; j += 256) {
            if (j < 384) {
                int g = j >> 7, h = j & 127;
                const float* W = (g == 0) ? timeW : (g == 1) ? attW : hintW;
                const float* gv = gateS + g * 128;
#pragma unroll
                for (int r = 0; r < 4; ++r) {
                    const float* Wr = W + r * 129 * HH;
                    float s = Wr[128 * HH + h];   // pad element = 1.0
#pragma unroll 8
                    for (int i2 = 0; i2 < 128; ++i2) s += gv[i2] * Wr[i2 * HH + h];
                    fus[(g * 4 + r) * 128 + h] = s;
                }
            } else {
                int h = j & 127;
                float s = gate3_b[h] + tF * wt[h] + aF * wa[h] + hF * wh[h];
#pragma unroll 8
                for (int k = 0; k < 128; ++k) s += interS[k] * gate3_W[(128 + k) * HH + h];
                baseS[h] = s;
            }
        }
        if (tid == 0) {
            float s = 0.f;
            for (int h = 0; h < 128; ++h) s += predv[h];
            out[idx] = s * (1.0f / 128.0f);
        }
        __syncthreads();

        // phase D: lg = relu(sum_r Wf[r] * prod_g fus + cell_bias)
        if (tid < 128) {
            float s = 0.f;
#pragma unroll
            for (int r = 0; r < 4; ++r)
                s += Wf[r] * fus[r * 128 + tid] * fus[(4 + r) * 128 + tid]
                           * fus[(8 + r) * 128 + tid];
            s += cell_bias[tid];
            lgS[tid] = s > 0.f ? s : 0.f;
        }
        __syncthreads();

        // phase E: forget = h_pre @ W1 (WMMA), then state update.
        // Wave w owns rows 16w..16w+15 (reads/writes only its own hA strip).
        v16bf afr[4];
#pragma unroll
        for (int kb = 0; kb < 4; ++kb)
            afr[kb] = *(const v16bf*)(hA + (((wave * 4 + kb) * 32 + lane) << 4));
        v8f acc[8];
        v8f z = {0.f, 0.f, 0.f, 0.f, 0.f, 0.f, 0.f, 0.f};
#pragma unroll
        for (int n = 0; n < 8; ++n) acc[n] = z;
#pragma unroll
        for (int n = 0; n < 8; ++n) {
#pragma unroll
            for (int kb = 0; kb < 4; ++kb) {
                v16bf bf = *(const v16bf*)(W1p + (((n * 4 + kb) * 32 + lane) << 4));
                acc[n] = __builtin_amdgcn_wmma_f32_16x16x32_bf16(
                    false, afr[kb], false, bf, (short)0, acc[n], false, false);
            }
        }
        int rofs = (lane >= 16) ? 8 : 0;
        int cl = lane & 15;
#pragma unroll
        for (int n = 0; n < 8; ++n) {
#pragma unroll
            for (int v = 0; v < 8; ++v) {
                int c = wave * 16 + v + rofs;
                int hc = n * 16 + cl;
                float f = acc[n][v] + baseS[hc];
                float hn = hS[c * HH + hc] * sigf(f) + corrS[c] * lgS[hc];
                hS[c * HH + hc] = hn;
                int kb = hc >> 5, r = hc & 31;
                int lg2 = (r >> 3) & 1;
                int e = (r & 7) + ((r >= 16) ? 8 : 0);
                hA[(((wave * 4 + kb) * 32 + ((c & 15) + 16 * lg2)) << 4) + e] = (__bf16)hn;
            }
        }
        __syncthreads();
    }
}

// ---------------------------------------------------------------------------
extern "C" void kernel_launch(void* const* d_in, const int* in_sizes, int n_in,
                              void* d_out, int out_size, void* d_ws, size_t ws_size,
                              hipStream_t stream)
{
    const int*   qseq      = (const int*)d_in[0];
    const int*   cseq      = (const int*)d_in[1];
    const float* tfv       = (const float*)d_in[2];
    const float* afv       = (const float*)d_in[3];
    const float* hfv       = (const float*)d_in[4];
    const float* E_q       = (const float*)d_in[5];
    const float* E_c       = (const float*)d_in[6];
    const float* q_matrix  = (const float*)d_in[7];
    const float* W_in      = (const float*)d_in[8];
    const float* b_in      = (const float*)d_in[9];
    const float* init_h    = (const float*)d_in[10];
    const float* tg_W      = (const float*)d_in[11];
    const float* tg_b      = (const float*)d_in[12];
    const float* ag_W      = (const float*)d_in[13];
    const float* ag_b      = (const float*)d_in[14];
    const float* hg_W      = (const float*)d_in[15];
    const float* hg_b      = (const float*)d_in[16];
    const float* timeW     = (const float*)d_in[17];
    const float* attW      = (const float*)d_in[18];
    const float* hintW     = (const float*)d_in[19];
    const float* Wf        = (const float*)d_in[20];
    const float* cell_bias = (const float*)d_in[21];
    const float* gate3_W   = (const float*)d_in[22];
    const float* gate3_b   = (const float*)d_in[23];
    const float* out_W     = (const float*)d_in[24];
    const float* out_b     = (const float*)d_in[25];
    float* out = (float*)d_out;

    char* ws = (char*)d_ws;
    float*  acts  = (float*)ws;                                   // 3200*128*4 = 1,638,400 B
    __bf16* winp  = (__bf16*)(ws + 1638400);                      // 65,536 B
    __bf16* w1p   = (__bf16*)(ws + 1638400 + 65536);              // 32,768 B
    float*  wsums = (float*)(ws + 1638400 + 65536 + 32768);       // 1,536 B

    lbkt_pack<<<1, 256, 0, stream>>>(W_in, gate3_W, winp, w1p, wsums);
    lbkt_acts<<<25, 256, 0, stream>>>(qseq, cseq, E_q, E_c, winp, b_in, acts);

    size_t smem = 65536 + 32768 + 32768 + 2816 * sizeof(float);   // 142,336 B
    lbkt_scan<<<BB, 256, smem, stream>>>(qseq, tfv, afv, hfv, E_q, q_matrix, init_h,
        tg_W, tg_b, ag_W, ag_b, hg_W, hg_b, timeW, attW, hintW, Wf, cell_bias,
        gate3_W, gate3_b, out_W, out_b, acts, w1p, wsums, out);
}
